// SNN_Autoencoder_71760313582244
// MI455X (gfx1250) — matrix operand order repaired and checked by reading, not compiled, not run on hardware
//
#include <hip/hip_runtime.h>
#include <cmath>

typedef __attribute__((ext_vector_type(16))) _Float16 v16h;
typedef __attribute__((ext_vector_type(8)))  _Float16 v8h;
typedef __attribute__((ext_vector_type(8)))  float    v8f;

#define THRESH 0.3f
#define LDS_STRIDE 40            // 32 halfs + 8 pad (80B rows, 16B aligned)
#define LDS_TILE   (128 * LDS_STRIDE)

// ---------------------------------------------------------------------------
// CDNA5 async global->LDS (B128 = 16 bytes/lane), tracked by ASYNCcnt.
// ---------------------------------------------------------------------------
__device__ __forceinline__ void async_b128_to_lds(unsigned ldsByteAddr, const void* gaddr) {
    asm volatile("global_load_async_to_lds_b128 %0, %1, off"
                 :: "v"(ldsByteAddr), "v"(gaddr)
                 : "memory");
}
__device__ __forceinline__ void wait_asynccnt0() {
    asm volatile("s_wait_asynccnt 0" ::: "memory");
}

// ---------------------------------------------------------------------------
// Zero-init (grid-stride over u32 words)
// ---------------------------------------------------------------------------
__global__ void zero_words_kernel(unsigned int* p, size_t nwords) {
    size_t i = (size_t)blockIdx.x * blockDim.x + threadIdx.x;
    size_t stride = (size_t)gridDim.x * blockDim.x;
    for (; i < nwords; i += stride) p[i] = 0u;
}

// ---------------------------------------------------------------------------
// Convert f32 weight [N,K] -> zero-padded f16 [Np,Kp] (row-major)
// ---------------------------------------------------------------------------
__global__ void convert_weight_kernel(const float* __restrict__ src,
                                      _Float16* __restrict__ dst,
                                      int N, int K, int Np, int Kp) {
    int idx = blockIdx.x * blockDim.x + threadIdx.x;
    int total = Np * Kp;
    if (idx >= total) return;
    int r = idx / Kp;
    int c = idx - r * Kp;
    float v = (r < N && c < K) ? src[(size_t)r * K + c] : 0.0f;
    dst[idx] = (_Float16)v;
}

// ---------------------------------------------------------------------------
// Fused GEMM (spikes @ W^T) + LIF membrane update, software-pipelined:
//   - double-buffered LDS A tile, staged via async global->LDS (f16 layers)
//   - next-step B fragments prefetched into registers during WMMA
// Block tile 128x64, 8 waves (4x2), wave tile 32x32 = 2x2 WMMA 16x16x32.
// ---------------------------------------------------------------------------
template<bool A_IS_F32, bool IS_OUT>
__global__ __launch_bounds__(256)
void lif_gemm_kernel(const void* __restrict__ Av, int aStride, int aValidK, int aValid,
                     const _Float16* __restrict__ W,       // padded [Np x Kp]
                     float* __restrict__ memB,             // [512 x Np]
                     _Float16* __restrict__ spkB,          // [512 x Np]
                     int Np, int Kp, float alpha,
                     float* __restrict__ out, int tOut)
{
    __shared__ __align__(16) _Float16 ldsA[2 * LDS_TILE];

    const int tid    = threadIdx.x;
    const int lane   = tid & 31;
    const int wave   = tid >> 5;
    const int wm     = wave >> 1;   // 0..3
    const int wn     = wave & 1;    // 0..1
    const int mBlock = blockIdx.y * 128;
    const int nBlock = blockIdx.x * 64;

    const int sRow  = tid >> 1;        // 0..127 staging row
    const int sHalf = (tid & 1) * 16;  // 0 or 16 (K offset within 32-step)

    // ---- A-tile staging: async for f16 spikes, sync+convert for f32 x ----
    auto stageA = [&](int bufSel, int k0) {
        _Float16* dst = &ldsA[bufSel * LDS_TILE + sRow * LDS_STRIDE + sHalf];
        if (A_IS_F32) {
            const float* arow = (const float*)Av +
                                (size_t)(mBlock + sRow) * aStride + k0 + sHalf;
            v8h h0, h1;
            if (aValid && (k0 + sHalf + 16 <= aValidK)) {       // fast path
                const float4* s4 = (const float4*)arow;
                float4 f0 = s4[0], f1 = s4[1], f2 = s4[2], f3 = s4[3];
                h0[0]=(_Float16)f0.x; h0[1]=(_Float16)f0.y; h0[2]=(_Float16)f0.z; h0[3]=(_Float16)f0.w;
                h0[4]=(_Float16)f1.x; h0[5]=(_Float16)f1.y; h0[6]=(_Float16)f1.z; h0[7]=(_Float16)f1.w;
                h1[0]=(_Float16)f2.x; h1[1]=(_Float16)f2.y; h1[2]=(_Float16)f2.z; h1[3]=(_Float16)f2.w;
                h1[4]=(_Float16)f3.x; h1[5]=(_Float16)f3.y; h1[6]=(_Float16)f3.z; h1[7]=(_Float16)f3.w;
            } else {                                            // boundary / t<5 only
                #pragma unroll
                for (int j = 0; j < 8; ++j) {
                    int k_lo = k0 + sHalf + j;
                    int k_hi = k_lo + 8;
                    h0[j] = (_Float16)((aValid && k_lo < aValidK) ? arow[j]     : 0.0f);
                    h1[j] = (_Float16)((aValid && k_hi < aValidK) ? arow[j + 8] : 0.0f);
                }
            }
            v8h* d = (v8h*)dst;
            d[0] = h0;
            d[1] = h1;
        } else {
            const _Float16* gsrc = (const _Float16*)Av +
                                   (size_t)(mBlock + sRow) * aStride + k0 + sHalf;
            unsigned ldsOff = (unsigned)(size_t)dst;
            async_b128_to_lds(ldsOff,      (const void*)gsrc);
            async_b128_to_lds(ldsOff + 16, (const void*)(gsrc + 8));
        }
    };

    // ---- B fragments straight from global (lane n holds W[col n, 16 contig K]) ----
    auto loadB = [&](int k0, v16h b[2]) {
        #pragma unroll
        for (int in = 0; in < 2; ++in) {
            const int col = nBlock + wn * 32 + in * 16 + (lane & 15);
            const _Float16* wptr = W + (size_t)col * Kp + k0 + (lane >> 4) * 16;
            b[in] = *(const v16h*)wptr;
        }
    };

    v8f acc[2][2] = {{{}, {}}, {{}, {}}};

    // prologue: stage first A tile, prefetch first B fragments
    stageA(0, 0);
    v16h bcur[2];
    loadB(0, bcur);

    int buf = 0;
    for (int k0 = 0; k0 < Kp; k0 += 32) {
        if (!A_IS_F32) wait_asynccnt0();      // our async writes to ldsA[buf] done
        __syncthreads();                       // publish to all waves; prior reads done
        const int next = k0 + 32;
        if (next < Kp) stageA(buf ^ 1, next);  // overlap next stage with this WMMA step

        // ---- A fragments from LDS (contiguous k order, as in B: lane<16 K{0..15}, lane>=16 K{16..31}) ----
        const int arow16 = lane & 15;
        const int koff   = (lane >> 4) * 16;
        const _Float16* lbase = &ldsA[buf * LDS_TILE];
        v16h afrag[2];
        #pragma unroll
        for (int im = 0; im < 2; ++im) {
            const int r  = wm * 32 + im * 16 + arow16;
            v8h lo = *(const v8h*)(&lbase[r * LDS_STRIDE + koff]);
            v8h hi = *(const v8h*)(&lbase[r * LDS_STRIDE + 8 + koff]);
            v16h a;
            #pragma unroll
            for (int j = 0; j < 8; ++j) { a[j] = lo[j]; a[8 + j] = hi[j]; }
            afrag[im] = a;
        }

        v16h bnxt[2];
        if (next < Kp) loadB(next, bnxt);      // prefetch next-step B during WMMA

        #pragma unroll
        for (int im = 0; im < 2; ++im)
            #pragma unroll
            for (int in = 0; in < 2; ++in)
                acc[im][in] = __builtin_amdgcn_wmma_f32_16x16x32_f16(
                    false, afrag[im], false, bcur[in],
                    (short)0, acc[im][in], false, false);

        if (next < Kp) { bcur[0] = bnxt[0]; bcur[1] = bnxt[1]; }
        buf ^= 1;
    }

    // ---- fused LIF epilogue (C/D layout: n = lane%16, m = vgpr + 8*(lane/16)) ----
    const int nLane = lane & 15;
    const int mOff  = (lane >> 4) * 8;
    #pragma unroll
    for (int im = 0; im < 2; ++im) {
        #pragma unroll
        for (int in = 0; in < 2; ++in) {
            const int nG = nBlock + wn * 32 + in * 16 + nLane;
            #pragma unroll
            for (int v = 0; v < 8; ++v) {
                const int    mG  = mBlock + wm * 32 + im * 16 + mOff + v;
                const size_t idx = (size_t)mG * Np + nG;
                float mem  = memB[idx];
                float sp   = (float)spkB[idx];
                float mnew = mem * alpha * (1.0f - sp) + acc[im][in][v];
                float snew = (mnew > THRESH) ? 1.0f : 0.0f;
                memB[idx] = (mnew < THRESH) ? mnew : 0.0f;
                spkB[idx] = (_Float16)snew;
                if (IS_OUT) {
                    if (tOut >= 0 && nG < 700)
                        out[((size_t)mG * 100 + tOut) * 700 + nG] = snew;
                }
            }
        }
    }
}

// ---------------------------------------------------------------------------
// Host launcher
// ---------------------------------------------------------------------------
extern "C" void kernel_launch(void* const* d_in, const int* in_sizes, int n_in,
                              void* d_out, int out_size, void* d_ws, size_t ws_size,
                              hipStream_t stream) {
    (void)in_sizes; (void)n_in; (void)out_size; (void)ws_size;

    const float* x = (const float*)d_in[0];
    const float* Wsrc[6] = {
        (const float*)d_in[1], (const float*)d_in[2], (const float*)d_in[3],
        (const float*)d_in[4], (const float*)d_in[5], (const float*)d_in[6]
    };
    // Layers: in->h1, h1->h2, h2->h3, h3->h2, h2->h1, h1->out
    const int N [6] = {512, 256, 128, 256, 512, 700};
    const int K [6] = {700, 512, 256, 128, 256, 512};
    const int Np[6] = {512, 256, 128, 256, 512, 704};  // mult of 64; Np[L]==Kp[L+1]
    const int Kp[6] = {704, 512, 256, 128, 256, 512};  // mult of 32

    char*  ws  = (char*)d_ws;
    size_t off = 0;
    auto carve = [&](size_t bytes) -> size_t {
        size_t o = off;
        off += (bytes + 255) & ~(size_t)255;
        return o;
    };

    _Float16* Wp  [6];
    float*    memB[6];
    _Float16* spkB[6];
    for (int L = 0; L < 6; ++L)
        Wp[L] = (_Float16*)(ws + carve((size_t)Np[L] * Kp[L] * sizeof(_Float16)));
    size_t stateBase = off;
    for (int L = 0; L < 6; ++L)
        memB[L] = (float*)(ws + carve((size_t)512 * Np[L] * sizeof(float)));
    for (int L = 0; L < 6; ++L)
        spkB[L] = (_Float16*)(ws + carve((size_t)512 * Np[L] * sizeof(_Float16)));
    size_t stateWords = (off - stateBase) / 4;

    // 1) zero LIF state (mem + spikes) — fresh carry every call
    zero_words_kernel<<<2048, 256, 0, stream>>>((unsigned int*)(ws + stateBase), stateWords);

    // 2) convert + pad weights to f16 (live in L2: ~2MB, reused 105x)
    for (int L = 0; L < 6; ++L) {
        int total = Np[L] * Kp[L];
        convert_weight_kernel<<<(total + 255) / 256, 256, 0, stream>>>(
            Wsrc[L], Wp[L], N[L], K[L], Np[L], Kp[L]);
    }

    // 3) time loop: 105 steps (5 zero-pad + 100), 6 dependent fused GEMM+LIF each
    const float alpha = expf(-1.0f / 3.0f);
    float* out = (float*)d_out;

    for (int t = 0; t < 105; ++t) {
        // layer 1: A = x[:, t-5, :] (f32 binary, row stride WIN*N_IN = 70000)
        const float* xt = (t >= 5) ? (x + (size_t)(t - 5) * 700) : x;
        lif_gemm_kernel<true, false><<<dim3(Np[0] / 64, 4), 256, 0, stream>>>(
            (const void*)xt, 70000, 700, (t >= 5) ? 1 : 0,
            Wp[0], memB[0], spkB[0], Np[0], Kp[0], alpha, nullptr, 0);

        // layers 2..5: A = previous layer's spikes (f16, padded, fully valid)
        for (int L = 1; L < 5; ++L) {
            lif_gemm_kernel<false, false><<<dim3(Np[L] / 64, 4), 256, 0, stream>>>(
                (const void*)spkB[L - 1], Kp[L], Kp[L], 1,
                Wp[L], memB[L], spkB[L], Np[L], Kp[L], alpha, nullptr, 0);
        }

        // layer 6: writes output spikes for t >= 5
        lif_gemm_kernel<false, true><<<dim3(11, 4), 256, 0, stream>>>(
            (const void*)spkB[4], Kp[5], Kp[5], 1,
            Wp[5], memB[5], spkB[5], Np[5], Kp[5], alpha, out, t - 5);
    }
}
